// GIN_63015760167232
// MI455X (gfx1250) — matrix operand order repaired and checked by reading, not compiled, not run on hardware
//
#include <hip/hip_runtime.h>

#define D   128
#define NN  50000
#define EE  800000
#define TM  128   // M rows per block

typedef _Float16 v16h __attribute__((ext_vector_type(16)));
typedef _Float16 v8h  __attribute__((ext_vector_type(8)));
typedef float    v8f  __attribute__((ext_vector_type(8)));

// ---------------- zero fill ----------------
__global__ void kzero(float* __restrict__ p, int n) {
    int i = blockIdx.x * blockDim.x + threadIdx.x;
    if (i < n) p[i] = 0.f;
}

// ---------------- edge scatter-add: agg[row[e]] += x[col[e]] ----------------
// one wave (32 lanes) per edge, float4 per lane -> 128 channels
__global__ void kscatter(const float* __restrict__ x,
                         const int* __restrict__ row,
                         const int* __restrict__ col,
                         float* __restrict__ agg) {
    int gtid = blockIdx.x * 256 + threadIdx.x;
    int e    = gtid >> 5;
    int lane = gtid & 31;
    if (e >= EE) return;
    int r = row[e];
    int c = col[e];
    const float4* xv = (const float4*)(x + (size_t)c * D);
    float4 v = xv[lane];
    float* dst = agg + (size_t)r * D + lane * 4;
    atomicAdd(dst + 0, v.x);
    atomicAdd(dst + 1, v.y);
    atomicAdd(dst + 2, v.z);
    atomicAdd(dst + 3, v.w);
}

// ---------------- elementwise BN affine + relu ----------------
__global__ void kaffine_relu(const float* __restrict__ in,
                             const float* __restrict__ scale,
                             const float* __restrict__ shift,
                             float* out, int n) {
    int i = blockIdx.x * blockDim.x + threadIdx.x;
    if (i >= n) return;
    int k = i & (D - 1);
    float v = in[i] * scale[k] + shift[k];
    out[i] = v > 0.f ? v : 0.f;
}

// ---------------- BN stats -> per-channel scale/shift; re-zero stats ----------------
__global__ void kbn_finalize(float* __restrict__ stats,          // [2*D] in: sums; zeroed on exit
                             const float* __restrict__ g,
                             const float* __restrict__ be,
                             float* __restrict__ scale,
                             float* __restrict__ shift) {
    int i = threadIdx.x;
    if (i < D) {
        float s  = stats[i];
        float s2 = stats[D + i];
        float m   = s / (float)NN;
        float var = s2 / (float)NN - m * m;
        float inv = rsqrtf(var + 1e-5f);
        float sc  = g[i] * inv;
        scale[i] = sc;
        shift[i] = be[i] - m * sc;
        stats[i] = 0.f;
        stats[D + i] = 0.f;
    }
}

// ---------------- fused pre-op + GEMM (WMMA f16->f32) + bias + BN stats ----------------
// mode 0: A = pre + agg        (GIN aggregation input)
// mode 1: A = relu(pre*scale + shift)   (BN affine + relu input)
// out = A @ W + bias ; if do_stats: stats[n] += sum_m out, stats[D+n] += sum_m out^2
__launch_bounds__(256)
__global__ void kgemm(const float* __restrict__ pre,
                      const float* __restrict__ agg,
                      const float* __restrict__ scale,
                      const float* __restrict__ shift,
                      const float* __restrict__ W,      // [K=D][N=D] row-major
                      const float* __restrict__ bias,   // [D]
                      float* __restrict__ out,
                      float* __restrict__ stats,        // [2*D] (atomics) or null
                      int mode, int do_stats) {
    __shared__ __align__(32) _Float16 lds_a[TM * D];  // A tile, [m][k], f16
    __shared__ __align__(32) _Float16 lds_w[D * D];   // W transposed, [n][k], f16

    const int tid = threadIdx.x;
    const int bm  = blockIdx.x * TM;

    // stage W^T (coalesced global read along n, f32 -> f16)
#pragma unroll
    for (int it = 0; it < (D * D) / 256; ++it) {
        int idx = it * 256 + tid;
        int k = idx >> 7;
        int n = idx & (D - 1);
        lds_w[n * D + k] = (_Float16)W[idx];
    }
    // stage A tile with fused pre-op
#pragma unroll
    for (int it = 0; it < (TM * D) / 256; ++it) {
        int idx = it * 256 + tid;
        int ml = idx >> 7;
        int k  = idx & (D - 1);
        int m  = bm + ml;
        float v = 0.f;
        if (m < NN) {
            size_t off = (size_t)m * D + k;
            if (mode == 0) {
                v = pre[off] + agg[off];
            } else {
                v = pre[off] * scale[k] + shift[k];
                v = v > 0.f ? v : 0.f;
            }
        }
        lds_a[ml * D + k] = (_Float16)v;
    }
    __syncthreads();

    const int wave = tid >> 5;        // 8 waves, each owns 16 M-rows
    const int lane = tid & 31;
    const int lm   = lane & 15;
    const int h    = lane >> 4;
    const int arow = wave * 16 + lm;

    v8f acc[8] = {};                  // 8 N-tiles of 16x16 f32

#pragma unroll
    for (int c = 0; c < 4; ++c) {     // K chunks of 32
        const int kb = c * 32;
        // A fragment: 16-bit A 16x32 layout (VGPR0-3: K=8h..8h+7, VGPR4-7: K=16+8h..)
        v8h lo = *(const v8h*)&lds_a[arow * D + kb + 8 * h];
        v8h hi = *(const v8h*)&lds_a[arow * D + kb + 16 + 8 * h];
        v16h afrag = __builtin_shufflevector(lo, hi,
                        0,1,2,3,4,5,6,7,8,9,10,11,12,13,14,15);
#pragma unroll
        for (int t = 0; t < 8; ++t) {
            int n = t * 16 + lm;
            // B fragment: same K layout as A (elems 0-7: K=8h..8h+7, elems 8-15: K=16+8h..)
            v8h blo = *(const v8h*)&lds_w[n * D + kb + 8 * h];
            v8h bhi = *(const v8h*)&lds_w[n * D + kb + 16 + 8 * h];
            v16h bfrag = __builtin_shufflevector(blo, bhi,
                        0,1,2,3,4,5,6,7,8,9,10,11,12,13,14,15);
            acc[t] = __builtin_amdgcn_wmma_f32_16x16x32_f16(
                         false, afrag, false, bfrag,
                         (short)0, acc[t], false, false);
        }
    }

    // epilogue: bias, store, per-column stats
#pragma unroll
    for (int t = 0; t < 8; ++t) {
        const int n = t * 16 + lm;
        const float bn = bias[n];
        float s = 0.f, s2 = 0.f;
#pragma unroll
        for (int j = 0; j < 8; ++j) {
            int m = bm + wave * 16 + 8 * h + j;   // C/D layout: VGPR j -> row j + 8h
            float v = acc[t][j] + bn;
            if (m < NN) {
                out[(size_t)m * D + n] = v;
                s  += v;
                s2 += v * v;
            }
        }
        if (do_stats) {
            s  += __shfl_xor(s, 16);
            s2 += __shfl_xor(s2, 16);
            if (h == 0) {
                atomicAdd(&stats[n], s);
                atomicAdd(&stats[D + n], s2);
            }
        }
    }
}

extern "C" void kernel_launch(void* const* d_in, const int* in_sizes, int n_in,
                              void* d_out, int out_size, void* d_ws, size_t ws_size,
                              hipStream_t stream) {
    const float* x      = (const float*)d_in[0];
    const int*   row    = (const int*)d_in[1];
    const int*   col    = (const int*)d_in[2];
    const float* c0_w1  = (const float*)d_in[3];
    const float* c0_b1  = (const float*)d_in[4];
    const float* c0_g   = (const float*)d_in[5];
    const float* c0_be  = (const float*)d_in[6];
    const float* c0_w2  = (const float*)d_in[7];
    const float* c0_b2  = (const float*)d_in[8];
    const float* bn0_g  = (const float*)d_in[9];
    const float* bn0_be = (const float*)d_in[10];
    const float* c1_w1  = (const float*)d_in[11];
    const float* c1_b1  = (const float*)d_in[12];
    const float* c1_g   = (const float*)d_in[13];
    const float* c1_be  = (const float*)d_in[14];
    const float* c1_w2  = (const float*)d_in[15];
    const float* c1_b2  = (const float*)d_in[16];
    float* out = (float*)d_out;

    char* ws = (char*)d_ws;
    float* stats = (float*)ws;            // 2*D
    float* scale = (float*)(ws + 1024);   // D
    float* shift = (float*)(ws + 1536);   // D
    const size_t ND = (size_t)NN * D;
    float* agg  = (float*)(ws + 2048);    // [N,D]
    float* bufB = agg + ND;               // h1b / h2b
    float* bufC = bufB + ND;              // h1d / h1e (in-place affine)

    const int ZB = 256;
    const int ndGrid   = (int)((ND + ZB - 1) / ZB);
    const int gemmGrid = (NN + TM - 1) / TM;
    const int scatGrid = (EE * 32 + 255) / 256;

    // ---- layer 0: gin_conv(x) ----
    kzero<<<ndGrid, ZB, 0, stream>>>(agg, (int)ND);
    kzero<<<1, 256, 0, stream>>>(stats, 2 * D);
    kscatter<<<scatGrid, 256, 0, stream>>>(x, row, col, agg);
    // h1b = (x+agg) @ w1 + b1, accumulate BN stats
    kgemm<<<gemmGrid, 256, 0, stream>>>(x, agg, nullptr, nullptr,
                                        c0_w1, c0_b1, bufB, stats, 0, 1);
    kbn_finalize<<<1, 128, 0, stream>>>(stats, c0_g, c0_be, scale, shift);
    // h1d = relu(bn(h1b)) @ w2 + b2, accumulate BN stats (outer bn0)
    kgemm<<<gemmGrid, 256, 0, stream>>>(bufB, nullptr, scale, shift,
                                        c0_w2, c0_b2, bufC, stats, 1, 1);
    kbn_finalize<<<1, 128, 0, stream>>>(stats, bn0_g, bn0_be, scale, shift);
    // h1e = relu(bn0(h1d))  (in-place, elementwise)
    kaffine_relu<<<ndGrid, ZB, 0, stream>>>(bufC, scale, shift, bufC, (int)ND);

    // ---- layer 1: gin_conv(h1e) ----
    kzero<<<ndGrid, ZB, 0, stream>>>(agg, (int)ND);
    kscatter<<<scatGrid, 256, 0, stream>>>(bufC, row, col, agg);
    kgemm<<<gemmGrid, 256, 0, stream>>>(bufC, agg, nullptr, nullptr,
                                        c1_w1, c1_b1, bufB, stats, 0, 1);
    kbn_finalize<<<1, 128, 0, stream>>>(stats, c1_g, c1_be, scale, shift);
    // out = relu(bn(h2b)) @ w2 + b2   (no stats)
    kgemm<<<gemmGrid, 256, 0, stream>>>(bufB, nullptr, scale, shift,
                                        c1_w2, c1_b2, out, nullptr, 1, 0);
}
